// LinearEdgeDecoder_37409165148918
// MI455X (gfx1250) — matrix operand order repaired and checked by reading, not compiled, not run on hardware
//
#include <hip/hip_runtime.h>

typedef __attribute__((ext_vector_type(16))) _Float16 v16h;
typedef __attribute__((ext_vector_type(8)))  _Float16 v8h;
typedef __attribute__((ext_vector_type(4)))  _Float16 v4h;
typedef __attribute__((ext_vector_type(8)))  float    v8f;

#define D_FEAT 128
#define H2     128

// ---------------- convert x: f32 -> f16 (halves gather traffic) ----------------
__global__ __launch_bounds__(256) void k_cvt_x(const float* __restrict__ x,
                                               _Float16* __restrict__ xh, int n) {
    int i = (blockIdx.x * 256 + threadIdx.x) * 4;
    if (i + 4 <= n) {
        const float4 v = *(const float4*)(x + i);
        v4h h;
        h[0] = (_Float16)v.x; h[1] = (_Float16)v.y;
        h[2] = (_Float16)v.z; h[3] = (_Float16)v.w;
        *(v4h*)(xh + i) = h;
    } else {
        for (; i < n; ++i) xh[i] = (_Float16)x[i];
    }
}

// ---- swizzle W1 (256x128 f32, row-major [K][N]) into WMMA B-fragment layout ----
// half index = ((kt*8 + nt)*32 + lane)*16 + e
//   B 32x16 f16 fragment: lane -> column N = nt*16 + (lane&15)
//                         K = kt*32 + (lane>=16 ? 8 : 0) + (e<8 ? e : e+8)
__global__ __launch_bounds__(256) void k_cvt_w1(const float* __restrict__ W1,
                                                _Float16* __restrict__ w1f) {
    int idx  = blockIdx.x * 256 + threadIdx.x;   // 0..32767
    int e    = idx & 15;
    int lane = (idx >> 4) & 31;
    int nt   = (idx >> 9) & 7;
    int kt   = idx >> 12;
    int K = kt * 32 + ((lane & 16) ? 8 : 0) + (e < 8 ? e : e + 8);
    int n = nt * 16 + (lane & 15);
    w1f[idx] = (_Float16)W1[K * H2 + n];
}

// ---------------- fused gather + GEMM(256->128) + ReLU + GEMV(128->1) ----------------
__global__ __launch_bounds__(256) void k_edge_mlp(
    const _Float16* __restrict__ xh,   // [N_NODES, 128] f16
    const _Float16* __restrict__ w1f,  // 32768 halves, fragment-swizzled W1
    const int*      __restrict__ eidx, // [2, nE]
    const float*    __restrict__ b1,   // [128]
    const float*    __restrict__ W2,   // [128]
    const float*    __restrict__ b2,   // [1]
    float*          __restrict__ out,  // [nE]
    int nE)
{
    __shared__ v16h w1s[2048];         // 64 fragments * 32 lanes = 64 KB

    const int tid = threadIdx.x;
    {   // stage W1 fragments into LDS (4096 x uint4)
        const uint4* s = (const uint4*)w1f;
        uint4*       d = (uint4*)w1s;
        #pragma unroll
        for (int i = 0; i < 16; ++i) d[tid + i * 256] = s[tid + i * 256];
    }
    __syncthreads();

    const int  lane = tid & 31;
    const int  wave = tid >> 5;
    const long long tile = (long long)blockIdx.x * 8 + wave;
    const long long tb   = tile * 16;
    if (tb >= nE) return;              // wave-uniform exit (EXEC stays all-1s for WMMA)

    const int ln = lane & 15;          // edge-in-tile / N column
    const int hs = lane >> 4;          // which K half this lane carries
    int e = (int)tb + ln;
    if (e >= nE) e = nE - 1;           // clamp gathers on the tail tile
    const int rn = eidx[e];
    const int cn = eidx[e + nE];
    const _Float16* pr = xh + (size_t)rn * D_FEAT;
    const _Float16* pc = xh + (size_t)cn * D_FEAT;
    const int hoff = hs * 8;

    v8f acc[8] = {};

    #pragma unroll
    for (int kt = 0; kt < 8; ++kt) {
        // A fragment 16x32 f16: elements 0..7 = K base..base+7, 8..15 = K base+16..base+23
        const _Float16* base = (kt < 4) ? (pr + kt * 32 + hoff)
                                        : (pc + (kt - 4) * 32 + hoff);
        v8h lo = *(const v8h*)(base);
        v8h hi = *(const v8h*)(base + 16);
        v16h a = __builtin_shufflevector(lo, hi, 0,1,2,3,4,5,6,7,8,9,10,11,12,13,14,15);
        #pragma unroll
        for (int nt = 0; nt < 8; ++nt) {
            v16h b = w1s[(kt * 8 + nt) * 32 + lane];
            acc[nt] = __builtin_amdgcn_wmma_f32_16x16x32_f16(
                false, a, false, b, (short)0, acc[nt], false, false);
        }
    }

    // Epilogue: bias + ReLU + dot(W2) fused in VALU.
    // C/D layout: lane -> N = nt*16 + ln ; VGPR i -> edge row M = i + hs*8
    const float b2v = b2[0];
    float bv[8], wv[8];
    #pragma unroll
    for (int nt = 0; nt < 8; ++nt) {
        int n = nt * 16 + ln;
        bv[nt] = b1[n];
        wv[nt] = W2[n];
    }
    float s[8];
    #pragma unroll
    for (int i = 0; i < 8; ++i) {
        float t = 0.f;
        #pragma unroll
        for (int nt = 0; nt < 8; ++nt) {
            float h = acc[nt][i] + bv[nt];
            h = h > 0.f ? h : 0.f;
            t += h * wv[nt];
        }
        s[i] = t;
    }
    // reduce over the 16 N-columns; xor masks 1..8 stay within each 16-lane half
    #pragma unroll
    for (int off = 8; off >= 1; off >>= 1) {
        #pragma unroll
        for (int i = 0; i < 8; ++i) s[i] += __shfl_xor(s[i], off, 32);
    }
    if (ln == 0) {
        const long long ebase = tb + (long long)hs * 8;
        #pragma unroll
        for (int i = 0; i < 8; ++i) {
            long long eo = ebase + i;
            if (eo < nE) out[eo] = s[i] + b2v;
        }
    }
}

extern "C" void kernel_launch(void* const* d_in, const int* in_sizes, int n_in,
                              void* d_out, int out_size, void* d_ws, size_t ws_size,
                              hipStream_t stream) {
    const float* x  = (const float*)d_in[0];
    const int*   ei = (const int*)d_in[1];   // JAX default x64-off -> int32
    const float* W1 = (const float*)d_in[2];
    const float* b1 = (const float*)d_in[3];
    const float* W2 = (const float*)d_in[4];
    const float* b2 = (const float*)d_in[5];
    float* out = (float*)d_out;

    const int nX = in_sizes[0];          // N_NODES * 128
    const int nE = in_sizes[1] / 2;

    _Float16* xh  = (_Float16*)d_ws;     // f16 copy of x
    _Float16* w1f = xh + (size_t)nX;     // 32768 halves of swizzled W1 (16B-aligned)

    k_cvt_x <<<(nX / 4 + 255) / 256, 256, 0, stream>>>(x, xh, nX);
    k_cvt_w1<<<128, 256, 0, stream>>>(W1, w1f);

    const long long tiles  = ((long long)nE + 15) / 16;
    const int       blocks = (int)((tiles + 7) / 8);
    k_edge_mlp<<<blocks, 256, 0, stream>>>(xh, w1f, ei, b1, W2, b2, out, nE);
}
